// Abstract_Loss_71665824301720
// MI455X (gfx1250) — matrix operand order repaired and checked by reading, not compiled, not run on hardware
//
#include <hip/hip_runtime.h>
#include <hip/hip_bf16.h>

typedef _Float16 half8  __attribute__((ext_vector_type(8)));
typedef _Float16 v16h   __attribute__((ext_vector_type(16)));
typedef float    v8f    __attribute__((ext_vector_type(8)));

#define BS 128
#define DDIM 768
#define NF 64
#define NC 32

// ---------------------------------------------------------------------------
// WMMA fragment loaders (contiguous k = 16h + s order in both A and B), operands row-major [rows, K],
// computing C = A * B^T  (B fragment is K x N, lane n holds column n's
// contiguous K values, which is exactly row n of the row-major B operand).
// ---------------------------------------------------------------------------
__device__ __forceinline__ v16h load_a_frag(const _Float16* __restrict__ A,
                                            int lda, int row0, int k0, int lane) {
  int l = lane & 15, h = lane >> 4;
  const _Float16* p = A + (size_t)(row0 + l) * lda + k0 + h * 16;
  return *(const v16h*)p;               // K = k0+h*16 .. +15
}

__device__ __forceinline__ v16h load_b_frag(const _Float16* __restrict__ B,
                                            int ldb, int col0, int k0, int lane) {
  int l = lane & 15, h = lane >> 4;
  const _Float16* p = B + (size_t)(col0 + l) * ldb + k0 + h * 16;
  return *(const v16h*)p;               // 16 contiguous K values
}

// ---------------------------------------------------------------------------
// Generic WMMA GEMM: C[M,N] = A[M,K] * B[N,K]^T, f16 in, f32 out.
// Block = 256 thr = 8 waves (4 m-tiles x 2 n-tiles) -> 64x32 C per block.
// grid = (N/32, M/64). K multiple of 32.
// ---------------------------------------------------------------------------
__global__ __launch_bounds__(256) void gemm_nt_wmma(
    const _Float16* __restrict__ A, const _Float16* __restrict__ B,
    float* __restrict__ C, int N, int K) {
  int lane = threadIdx.x & 31, wid = threadIdx.x >> 5;
  int m0 = blockIdx.y * 64 + (wid >> 1) * 16;
  int n0 = blockIdx.x * 32 + (wid & 1) * 16;
  v8f acc = {};
  for (int k0 = 0; k0 < K; k0 += 32) {
    v16h a = load_a_frag(A, K, m0, k0, lane);
    v16h b = load_b_frag(B, K, n0, k0, lane);
    acc = __builtin_amdgcn_wmma_f32_16x16x32_f16(false, a, false, b,
                                                 (short)0, acc, false, false);
  }
  int l = lane & 15, h = lane >> 4;
  for (int r = 0; r < 8; ++r)
    C[(size_t)(m0 + r + h * 8) * N + n0 + l] = acc[r];
}

// ---------------------------------------------------------------------------
// Fused per-pair (b,g) kernel: WMMA 64x32 sim tile -> LDS -> soft-DTW
// wavefront DP (wave 0, lanes = 32 columns, 95 anti-diagonals).
// grid = 16384 (= b*128 + g), block = 256.
// ---------------------------------------------------------------------------
__global__ __launch_bounds__(256) void dtw_fused(
    const _Float16* __restrict__ Vf, const _Float16* __restrict__ Tc,
    float* __restrict__ Dmat) {
  __shared__ float tile[NF * NC];   // 8 KB
  int lane = threadIdx.x & 31, wid = threadIdx.x >> 5;
  int p = blockIdx.x, b = p >> 7, g = p & 127;
  int m0 = (wid >> 1) * 16, n0 = (wid & 1) * 16;
  const _Float16* A = Vf + (size_t)b * NF * DDIM;
  const _Float16* B = Tc + (size_t)g * NC * DDIM;

  v8f acc = {};
  for (int k0 = 0; k0 < DDIM; k0 += 32) {
    v16h a = load_a_frag(A, DDIM, m0, k0, lane);
    v16h bf = load_b_frag(B, DDIM, n0, k0, lane);
    acc = __builtin_amdgcn_wmma_f32_16x16x32_f16(false, a, false, bf,
                                                 (short)0, acc, false, false);
  }
  int l = lane & 15, h = lane >> 4;
  for (int r = 0; r < 8; ++r)
    tile[(m0 + r + h * 8) * NC + n0 + l] = acc[r];
  __syncthreads();

  if (wid == 0) {
    const float beta = 0.1f, invb = 10.0f;
    int j = lane;
    float rm1 = 0.0f, rm2 = 0.0f;       // R at diag d-1 / d-2 for this column
    for (int d = 0; d < NF + NC - 1; ++d) {
      float left = __shfl_up(rm1, 1);   // R[i][j-1]
      float dgv  = __shfl_up(rm2, 1);   // R[i-1][j-1]
      float up   = rm1;                 // R[i-1][j]
      int i = d - j;
      if (i >= 0 && i < NF) {
        float c = 1.0f - tile[i * NC + j];
        float r;
        if (i == 0)      r = (j == 0) ? c : (left + c);   // row-0 cumsum
        else if (j == 0) r = up + c;                      // col-0 only-up
        else {
          float m = fminf(left, fminf(up, dgv));
          float s = __expf((m - left) * invb) + __expf((m - up) * invb) +
                    __expf((m - dgv) * invb);
          r = c + m - beta * __logf(s);                   // softmin3
        }
        rm2 = rm1; rm1 = r;
      }
    }
    if (j == 31) Dmat[p] = rm1;         // R[63][31]
  }
}

// ---------------------------------------------------------------------------
// Row L2-normalize (768 cols) f32 -> f16.  grid = nrows, block = 256.
// ---------------------------------------------------------------------------
__global__ __launch_bounds__(256) void rows_l2norm_f16(
    const float* __restrict__ in, _Float16* __restrict__ out) {
  __shared__ float sdata[8];
  int row = blockIdx.x;
  const float* rp = in + (size_t)row * DDIM;
  float v[3], ss = 0.0f;
  for (int t = 0; t < 3; ++t) {
    v[t] = rp[t * 256 + threadIdx.x];
    ss += v[t] * v[t];
  }
  for (int o = 16; o > 0; o >>= 1) ss += __shfl_xor(ss, o);
  int lane = threadIdx.x & 31, wid = threadIdx.x >> 5;
  if (lane == 0) sdata[wid] = ss;
  __syncthreads();
  float tot = 0.0f;
  for (int w = 0; w < 8; ++w) tot += sdata[w];
  float rn = 1.0f / fmaxf(sqrtf(tot), 1e-12f);
  _Float16* op = out + (size_t)row * DDIM;
  for (int t = 0; t < 3; ++t)
    op[t * 256 + threadIdx.x] = (_Float16)(v[t] * rn);
}

// mean over NC candidates then L2-normalize.  grid = 128, block = 256.
__global__ __launch_bounds__(256) void agg_l2norm_f16(
    const float* __restrict__ cand, _Float16* __restrict__ out) {
  __shared__ float sdata[8];
  int row = blockIdx.x;
  const float* base = cand + (size_t)row * NC * DDIM;
  float v[3], ss = 0.0f;
  for (int t = 0; t < 3; ++t) {
    int c = t * 256 + threadIdx.x;
    float s = 0.0f;
    for (int cc = 0; cc < NC; ++cc) s += base[(size_t)cc * DDIM + c];
    v[t] = s * (1.0f / (float)NC);
    ss += v[t] * v[t];
  }
  for (int o = 16; o > 0; o >>= 1) ss += __shfl_xor(ss, o);
  int lane = threadIdx.x & 31, wid = threadIdx.x >> 5;
  if (lane == 0) sdata[wid] = ss;
  __syncthreads();
  float tot = 0.0f;
  for (int w = 0; w < 8; ++w) tot += sdata[w];
  float rn = 1.0f / fmaxf(sqrtf(tot), 1e-12f);
  _Float16* op = out + (size_t)row * DDIM;
  for (int t = 0; t < 3; ++t)
    op[t * 256 + threadIdx.x] = (_Float16)(v[t] * rn);
}

// ---------------------------------------------------------------------------
// Diagonal-target cross entropy on logits = scale * L (optionally L^T).
// Accumulates weight * (-mean_i lsm[i,i]) into *acc.  1 wave per row.
// ---------------------------------------------------------------------------
__global__ __launch_bounds__(256) void ce_diag(
    const float* __restrict__ L, int n, float scale, float weight,
    int transposed, float* __restrict__ acc) {
  int lane = threadIdx.x & 31, wid = threadIdx.x >> 5;
  int i = blockIdx.x * 8 + wid;
  if (i >= n) return;
  float m = -3.4e38f;
  for (int j = lane; j < n; j += 32) {
    float x = scale * (transposed ? L[(size_t)j * n + i] : L[(size_t)i * n + j]);
    m = fmaxf(m, x);
  }
  for (int o = 16; o > 0; o >>= 1) m = fmaxf(m, __shfl_xor(m, o));
  float se = 0.0f;
  for (int j = lane; j < n; j += 32) {
    float x = scale * (transposed ? L[(size_t)j * n + i] : L[(size_t)i * n + j]);
    se += __expf(x - m);
  }
  for (int o = 16; o > 0; o >>= 1) se += __shfl_xor(se, o);
  if (lane == 0) {
    float xd = scale * L[(size_t)i * n + i];
    float lsm = xd - m - __logf(se);
    atomicAdd(acc, -lsm * weight / (float)n);
  }
}

// NT-Xent on raw z@z^T (256x256): temp 0.1, diag masked to -1e9,
// pos(i) = i^128 (i+128 / i-128).  Each row adds alpha/(n*5) * (-lsm_pos).
__global__ __launch_bounds__(256) void ntxent_rows(
    const float* __restrict__ S, float* __restrict__ acc) {
  int lane = threadIdx.x & 31, wid = threadIdx.x >> 5;
  int i = blockIdx.x * 8 + wid;           // 0..255
  float m = -3.4e38f;
  for (int j = lane; j < 256; j += 32) {
    float x = (j == i) ? -1e9f : S[(size_t)i * 256 + j] * 10.0f;
    m = fmaxf(m, x);
  }
  for (int o = 16; o > 0; o >>= 1) m = fmaxf(m, __shfl_xor(m, o));
  float se = 0.0f;
  for (int j = lane; j < 256; j += 32) {
    float x = (j == i) ? -1e9f : S[(size_t)i * 256 + j] * 10.0f;
    se += __expf(x - m);
  }
  for (int o = 16; o > 0; o >>= 1) se += __shfl_xor(se, o);
  if (lane == 0) {
    int pj = (i < 128) ? i + 128 : i - 128;
    float xp = S[(size_t)i * 256 + pj] * 10.0f;
    float lsm = xp - m - __logf(se);
    atomicAdd(acc, -lsm * (0.5f / (128.0f * 5.0f)));
  }
}

__global__ void init_acc(float* acc) { acc[0] = 0.0f; }
__global__ void finalize(const float* acc, float* out) { out[0] = acc[0]; }

// ---------------------------------------------------------------------------
extern "C" void kernel_launch(void* const* d_in, const int* in_sizes, int n_in,
                              void* d_out, int out_size, void* d_ws, size_t ws_size,
                              hipStream_t stream) {
  const float* video_embd       = (const float*)d_in[0];   // [128,768]
  const float* text_embd_sent   = (const float*)d_in[1];   // [128,768]
  const float* text_embd_cand   = (const float*)d_in[2];   // [128,32,768]
  const float* video_embd_frame = (const float*)d_in[3];   // [128,64,768]
  // d_in[4] = pos_step : unused by the reference computation.

  char* w = (char*)d_ws;
  float*     acc  = (float*)w;            w += 256;
  _Float16*  z16  = (_Float16*)w;         w += (size_t)256 * DDIM * 2;   // [v_n; t_n]
  _Float16*  An16 = (_Float16*)w;         w += (size_t)BS * DDIM * 2;
  _Float16*  Vf16 = (_Float16*)w;         w += (size_t)BS * NF * DDIM * 2;
  _Float16*  Tc16 = (_Float16*)w;         w += (size_t)BS * NC * DDIM * 2;
  float*     SimZ = (float*)w;            w += (size_t)256 * 256 * 4;
  float*     L1   = (float*)w;            w += (size_t)BS * BS * 4;
  float*     L3   = (float*)w;            w += (size_t)BS * BS * 4;
  float*     L4   = (float*)w;            w += (size_t)BS * BS * 4;
  float*     Dmat = (float*)w;            w += (size_t)BS * BS * 4;

  init_acc<<<1, 1, 0, stream>>>(acc);

  // Normalizations (f32 -> unit-norm f16)
  rows_l2norm_f16<<<BS, 256, 0, stream>>>(video_embd, z16);
  rows_l2norm_f16<<<BS, 256, 0, stream>>>(text_embd_sent, z16 + (size_t)BS * DDIM);
  agg_l2norm_f16 <<<BS, 256, 0, stream>>>(text_embd_cand, An16);
  rows_l2norm_f16<<<BS * NF, 256, 0, stream>>>(video_embd_frame, Vf16);
  rows_l2norm_f16<<<BS * NC, 256, 0, stream>>>(text_embd_cand, Tc16);

  // Logit GEMMs (WMMA): C = A * B^T
  gemm_nt_wmma<<<dim3(256 / 32, 256 / 64), 256, 0, stream>>>(z16, z16, SimZ, 256, DDIM);
  gemm_nt_wmma<<<dim3(BS / 32, BS / 64), 256, 0, stream>>>(z16, z16 + (size_t)BS * DDIM, L1, BS, DDIM);
  gemm_nt_wmma<<<dim3(BS / 32, BS / 64), 256, 0, stream>>>(z16, An16, L3, BS, DDIM);
  gemm_nt_wmma<<<dim3(BS / 32, BS / 64), 256, 0, stream>>>(z16 + (size_t)BS * DDIM, An16, L4, BS, DDIM);

  // Fused big GEMM + soft-DTW: one block per (b,g)
  dtw_fused<<<BS * BS, 256, 0, stream>>>(Vf16, Tc16, Dmat);

  // CLIP losses: 0.5*(ce + ce^T), each /5 -> weight 0.1 per direction
  ce_diag<<<BS / 8, 256, 0, stream>>>(L1, BS,  10.0f, 0.1f, 0, acc);
  ce_diag<<<BS / 8, 256, 0, stream>>>(L1, BS,  10.0f, 0.1f, 1, acc);
  ce_diag<<<BS / 8, 256, 0, stream>>>(L3, BS,  10.0f, 0.1f, 0, acc);
  ce_diag<<<BS / 8, 256, 0, stream>>>(L3, BS,  10.0f, 0.1f, 1, acc);
  ce_diag<<<BS / 8, 256, 0, stream>>>(L4, BS,  10.0f, 0.1f, 0, acc);
  ce_diag<<<BS / 8, 256, 0, stream>>>(L4, BS,  10.0f, 0.1f, 1, acc);

  // NT-Xent over 2N views
  ntxent_rows<<<256 / 8, 256, 0, stream>>>(SimZ, acc);

  // DTW contrastive: logits = -10*D, weight = 0.1 (REG_W) * 0.5 / 5 = 0.01
  ce_diag<<<BS / 8, 256, 0, stream>>>(Dmat, BS, -10.0f, 0.01f, 0, acc);
  ce_diag<<<BS / 8, 256, 0, stream>>>(Dmat, BS, -10.0f, 0.01f, 1, acc);

  finalize<<<1, 1, 0, stream>>>(acc, (float*)d_out);
}
